// BinaryDense_42167988912830
// MI455X (gfx1250) — matrix operand order repaired and checked by reading, not compiled, not run on hardware
//
#include <hip/hip_runtime.h>
#include <hip/hip_bf16.h>

typedef __bf16 bf16_t;
typedef __attribute__((ext_vector_type(16))) __bf16 v16bf;
typedef __attribute__((ext_vector_type(8)))  float  v8f;
typedef __attribute__((ext_vector_type(4)))  float  v4f;   // clang vector (nontemporal-OK)

#define TILE_M 128
#define TILE_N 256
#define TILE_K 32
#define APITCH 40    // 32 + 8 pad (bf16): 80B row pitch (16B aligned)
#define BPITCH 264   // 256 + 8 pad (bf16): 528B row pitch (16B aligned)

// 16B async copy global -> LDS (ASYNCcnt-tracked, no VGPR staging)
#define ASYNC_LD_B128(ldsoff, gptr)                                        \
    asm volatile("global_load_async_to_lds_b128 %0, %1, off"               \
                 :: "v"(ldsoff), "v"(gptr) : "memory")
#define WAIT_ASYNC0() asm volatile("s_wait_asynccnt 0" ::: "memory")

// ---------------------------------------------------------------------------
// Elementwise converters. Inputs are streamed once -> non-temporal loads
// (don't pollute L2); bf16 outputs are the GEMM working set (96 MB, fits in
// 192 MB L2) -> default RT stores so they stay resident.
// ---------------------------------------------------------------------------
__global__ __launch_bounds__(256)
void cvt_f32_to_bf16(const float* __restrict__ src, bf16_t* __restrict__ dst, long n) {
    long i = ((long)blockIdx.x * blockDim.x + threadIdx.x) * 4;
    if (i >= n) return;
    v4f v = __builtin_nontemporal_load((const v4f*)(src + i));
    union { bf16_t h[4]; unsigned long long u; } o;
    o.h[0] = (bf16_t)v.x; o.h[1] = (bf16_t)v.y;
    o.h[2] = (bf16_t)v.z; o.h[3] = (bf16_t)v.w;
    *(unsigned long long*)(dst + i) = o.u;
}

__global__ __launch_bounds__(256)
void sign_f32_to_bf16(const float* __restrict__ src, bf16_t* __restrict__ dst, long n) {
    long i = ((long)blockIdx.x * blockDim.x + threadIdx.x) * 4;
    if (i >= n) return;
    v4f v = __builtin_nontemporal_load((const v4f*)(src + i));
    union { bf16_t h[4]; unsigned long long u; } o;
    o.h[0] = (bf16_t)(float)((v.x > 0.f) - (v.x < 0.f));
    o.h[1] = (bf16_t)(float)((v.y > 0.f) - (v.y < 0.f));
    o.h[2] = (bf16_t)(float)((v.z > 0.f) - (v.z < 0.f));
    o.h[3] = (bf16_t)(float)((v.w > 0.f) - (v.w < 0.f));
    *(unsigned long long*)(dst + i) = o.u;
}

// ---------------------------------------------------------------------------
// Double-buffered async-LDS bf16 WMMA GEMM: C[M,N] = A[M,K]*B[K,N] + bias
// 256 threads = 8 waves (wave32). Workgroup tile 128x256, K-step 32.
// Wave grid 2(M) x 4(N); each wave computes 64x64 = 4x4 tiles of 16x16.
// K-loop unrolled by 2 so the ping-pong buffer index is compile-time constant.
// ---------------------------------------------------------------------------
__global__ __launch_bounds__(256)
void bdense_wmma(const bf16_t* __restrict__ A, const bf16_t* __restrict__ Bm,
                 const float* __restrict__ bias, float* __restrict__ C,
                 int M, int N, int K) {
    __shared__ __align__(16) bf16_t lA[2][TILE_M * APITCH];
    __shared__ __align__(16) bf16_t lB[2][TILE_K * BPITCH];

    const int tid  = threadIdx.x;
    const int lane = tid & 31;
    const int wave = tid >> 5;
    const int wm   = wave & 1;   // wave M index (0..1)
    const int wn   = wave >> 1;  // wave N index (0..3)
    const int half = lane >> 4;
    const int l16  = lane & 15;

    const long rowA0 = (long)blockIdx.x * TILE_M;
    const long colB0 = (long)blockIdx.y * TILE_N;

    // --- per-thread async-copy assignments -------------------------------
    // A tile: 128x32 = 4096 bf16, 2 chunks of 8 per thread
    // B tile:  32x256 = 8192 bf16, 4 chunks of 8 per thread
    const bf16_t* gA[2];
    const bf16_t* gB[4];
    unsigned lAoff0[2], lAoff1[2], lBoff0[4], lBoff1[4];
#pragma unroll
    for (int p = 0; p < 2; ++p) {
        int q = p * 256 + tid;
        int r = q >> 2, c = (q & 3) * 8;
        gA[p] = A + (rowA0 + r) * (long)K + c;
        lAoff0[p] = (unsigned)(unsigned long long)(const void*)(&lA[0][r * APITCH + c]);
        lAoff1[p] = (unsigned)(unsigned long long)(const void*)(&lA[1][r * APITCH + c]);
    }
#pragma unroll
    for (int p = 0; p < 4; ++p) {
        int q = p * 256 + tid;
        int r = q >> 5, c = (q & 31) * 8;
        gB[p] = Bm + (long)r * N + colB0 + c;
        lBoff0[p] = (unsigned)(unsigned long long)(const void*)(&lB[0][r * BPITCH + c]);
        lBoff1[p] = (unsigned)(unsigned long long)(const void*)(&lB[1][r * BPITCH + c]);
    }

    v8f acc[4][4];
#pragma unroll
    for (int mi = 0; mi < 4; ++mi)
#pragma unroll
        for (int ni = 0; ni < 4; ++ni)
            acc[mi][ni] = (v8f){0.f,0.f,0.f,0.f,0.f,0.f,0.f,0.f};

    const int nsteps = K / TILE_K;   // even (K multiple of 64)

    // stage K-step `STEP` into buffer set (AOFF,BOFF)
#define STAGE(AOFF, BOFF, STEP)                                              \
    do {                                                                     \
        const long ka_ = (long)(STEP) * TILE_K;                              \
        const long kb_ = (long)(STEP) * TILE_K * N;                          \
        _Pragma("unroll")                                                    \
        for (int p = 0; p < 2; ++p) ASYNC_LD_B128(AOFF[p], gA[p] + ka_);     \
        _Pragma("unroll")                                                    \
        for (int p = 0; p < 4; ++p) ASYNC_LD_B128(BOFF[p], gB[p] + kb_);     \
    } while (0)

    // consume LDS buffer BUF (compile-time constant): 16 WMMAs
#define COMPUTE(BUF)                                                         \
    do {                                                                     \
        v16bf afrag[4], bfrag[4];                                            \
        _Pragma("unroll")                                                    \
        for (int mi = 0; mi < 4; ++mi) {                                     \
            int row = wm * 64 + mi * 16 + l16;                               \
            union { uint4 u[2]; v16bf v; } t;                                \
            t.u[0] = *(const uint4*)(&lA[BUF][row * APITCH + half * 8]);     \
            t.u[1] = *(const uint4*)(&lA[BUF][row * APITCH + 16 + half * 8]);\
            afrag[mi] = t.v;                                                 \
        }                                                                    \
        _Pragma("unroll")                                                    \
        for (int ni = 0; ni < 4; ++ni) {                                     \
            int col = wn * 64 + ni * 16;                                     \
            union { uint4 u[2]; v16bf v; } t;                                \
            _Pragma("unroll")                                                \
            for (int s = 0; s < 16; ++s)                                     \
                t.v[s] = lB[BUF][((s < 8) ? (half * 8 + s)                   \
                                          : (16 + half * 8 + s - 8)) * BPITCH \
                                 + col + l16];                               \
            bfrag[ni] = t.v;                                                 \
        }                                                                    \
        _Pragma("unroll")                                                    \
        for (int mi = 0; mi < 4; ++mi)                                       \
            _Pragma("unroll")                                                \
            for (int ni = 0; ni < 4; ++ni)                                   \
                acc[mi][ni] = __builtin_amdgcn_wmma_f32_16x16x32_bf16(       \
                    false, afrag[mi], false, bfrag[ni],                      \
                    (short)0, acc[mi][ni], false, false);                    \
    } while (0)

    // prologue: stage first K-tile into buffer 0
    STAGE(lAoff0, lBoff0, 0);

    for (int i = 0; i < nsteps; i += 2) {
        // ---- even step: compute buf0, prefetch into buf1 ----
        WAIT_ASYNC0();       // this wave's loads into buf0 complete
        __syncthreads();     // all waves loaded buf0; all prior reads of buf1 done
        if (i + 1 < nsteps) STAGE(lAoff1, lBoff1, i + 1);
        COMPUTE(0);

        // ---- odd step: compute buf1, prefetch into buf0 ----
        WAIT_ASYNC0();
        __syncthreads();
        if (i + 2 < nsteps) STAGE(lAoff0, lBoff0, i + 2);
        COMPUTE(1);
    }
#undef STAGE
#undef COMPUTE

    // --- epilogue: C/D layout lane -> N=l16, VGPR r -> M = r + 8*half.
    // C is streamed once -> non-temporal stores keep A/B resident in L2.
#pragma unroll
    for (int mi = 0; mi < 4; ++mi) {
#pragma unroll
        for (int ni = 0; ni < 4; ++ni) {
            long col = colB0 + wn * 64 + ni * 16 + l16;
            float bv = bias[col];
#pragma unroll
            for (int r = 0; r < 8; ++r) {
                long row = rowA0 + wm * 64 + mi * 16 + half * 8 + r;
                __builtin_nontemporal_store(acc[mi][ni][r] + bv,
                                            C + row * (long)N + col);
            }
        }
    }
}

// ---------------------------------------------------------------------------
extern "C" void kernel_launch(void* const* d_in, const int* in_sizes, int n_in,
                              void* d_out, int out_size, void* d_ws, size_t ws_size,
                              hipStream_t stream) {
    const float* x    = (const float*)d_in[0];  // [B, D_IN] f32
    const float* w    = (const float*)d_in[1];  // [D_IN, UNITS] f32
    const float* bias = (const float*)d_in[2];  // [UNITS] f32
    float* out = (float*)d_out;                 // [B, UNITS] f32

    const long UNITS = in_sizes[2];
    const long DIN   = in_sizes[1] / UNITS;
    const long Bdim  = in_sizes[0] / DIN;

    bf16_t* Abf = (bf16_t*)d_ws;                // 64 MB
    bf16_t* Bbf = Abf + Bdim * DIN;             // 32 MB

    const long nx = Bdim * DIN;
    const long nw = DIN * UNITS;
    cvt_f32_to_bf16 <<<(unsigned)((nx / 4 + 255) / 256), 256, 0, stream>>>(x, Abf, nx);
    sign_f32_to_bf16<<<(unsigned)((nw / 4 + 255) / 256), 256, 0, stream>>>(w, Bbf, nw);

    dim3 grid((unsigned)(Bdim / TILE_M), (unsigned)(UNITS / TILE_N));
    bdense_wmma<<<grid, 256, 0, stream>>>(Abf, Bbf, bias, out,
                                          (int)Bdim, (int)UNITS, (int)DIN);
}
